// LSTM_30940944400986
// MI455X (gfx1250) — matrix operand order repaired and checked by reading, not compiled, not run on hardware
//
#include <hip/hip_runtime.h>

// ---------------------------------------------------------------------------
// LSTM on MI455X (gfx1250): bf16 WMMA GEMM per timestep, L2-resident weights.
//
// Workspace layout (bytes, needs ~4.2 MB):
//   [0        , 2097152)  Wh_sw  : W_h  [2048x512] bf16, A-operand swizzled
//   [2097152  , 2228224)  Wph_sw : W_ph [ 128x512] bf16, A-operand swizzled
//   [2228224  , 2752512)  h0     : h ping buffer, bf16, B-operand swizzled
//   [2752512  , 3276800)  h1     : h pong buffer
//   [3276800  , 4325376)  c      : cell state f32 [512x512]
// ---------------------------------------------------------------------------

typedef __bf16 v16bf __attribute__((ext_vector_type(16)));
typedef float  v8f   __attribute__((ext_vector_type(8)));

__device__ __forceinline__ unsigned short f32_to_bf16(float f) {
    unsigned int u = __float_as_uint(f);
    u += 0x7fffu + ((u >> 16) & 1u);   // round to nearest even
    return (unsigned short)(u >> 16);
}

// A-operand swizzle for V_WMMA_*_16X16X32_BF16 (ISA 7.12.2, 16-bit A 16x32):
// lanes 0-15 hold M=lane, K: VGPR0..3 -> K=0..7 (2/VGPR), VGPR4..7 -> K=16..23
// lanes 16-31 same M, K+8 / K+24.
__global__ __launch_bounds__(256) void convert_swizzle_A(
    const float* __restrict__ W, unsigned short* __restrict__ out, int rowBlocks)
{
    int idx = blockIdx.x * 256 + threadIdx.x;
    int total = rowBlocks * 16 * 512;   // rowBlocks * 16 kb-blocks * 512 elems
    if (idx >= total) return;
    int within = idx & 511;
    int blk    = idx >> 9;
    int kb     = blk & 15;              // K block (32 wide), K=512 fixed
    int ablk   = blk >> 4;              // 16-row block
    int lane   = within >> 4;
    int j      = within & 15;
    int M      = lane & 15;
    int laneHi = lane >> 4;
    int v = j >> 1, pos = j & 1;
    int k_local = (v < 4 ? 2 * v + pos : 16 + 2 * (v - 4) + pos) + 8 * laneHi;
    int row = ablk * 16 + M;
    int k   = kb * 32 + k_local;
    out[idx] = f32_to_bf16(W[row * 512 + k]);
}

// One LSTM timestep. Grid (32, 16): blockIdx.x = hidden block (16 rows),
// blockIdx.y = batch block (32 cols). 8 waves: wave = gate*2 + colSubtile.
__global__ __launch_bounds__(256) void lstm_step(
    const unsigned short* __restrict__ Wh_sw,
    const float* __restrict__ x,
    const float* __restrict__ W_x,
    const float* __restrict__ bg, const float* __restrict__ bi,
    const float* __restrict__ bfv, const float* __restrict__ bo,
    const unsigned short* __restrict__ h_src,
    unsigned short* __restrict__ h_dst,
    float* __restrict__ c, int t)
{
    __shared__ float gatesLds[4 * 16 * 32];   // 8 KB

    const int hb   = blockIdx.x;          // 0..31
    const int nb2  = blockIdx.y;          // 0..15 (32 batch cols each)
    const int wave = threadIdx.x >> 5;
    const int lane = threadIdx.x & 31;
    const int g    = wave >> 1;           // gate 0..3  (g,i,f,o)
    const int s    = wave & 1;            // 16-col subtile
    const int nblk = nb2 * 2 + s;         // global 16-col block 0..31
    const int ablk = g * 32 + hb;         // row-block index in 4H dim

    const unsigned short* aPtr = Wh_sw + (ablk * 16) * 512 + lane * 16;
    const unsigned short* bPtr = h_src + nblk * 512 + lane * 16;

    v8f acc = {};
#pragma unroll
    for (int kb = 0; kb < 16; ++kb) {     // K = 512 = 16 x 32
        v16bf a = *reinterpret_cast<const v16bf*>(aPtr + kb * 512);
        v16bf b = *reinterpret_cast<const v16bf*>(bPtr + kb * (32 * 512));
        acc = __builtin_amdgcn_wmma_f32_16x16x32_bf16(
            false, a, false, b, (short)0, acc, false, false);
    }

    union { v8f v; float f[8]; } u;
    u.v = acc;

    // C/D layout: N = lane%16 (+block), M = r + 8*(lane/16)
    const int nl = s * 16 + (lane & 15);          // column within 32-col block
    const int n  = nb2 * 32 + nl;                 // global batch column
    const float xt = x[n * 128 + t];              // x is [B, SEQ_LEN]
    const float* bias = (g == 0) ? bg : (g == 1) ? bi : (g == 2) ? bfv : bo;

#pragma unroll
    for (int r = 0; r < 8; ++r) {
        int M  = r + 8 * (lane >> 4);
        int hr = hb * 16 + M;                     // hidden row 0..511
        float val = u.f[r] + W_x[g * 512 + hr] * xt + bias[hr];
        val = (g == 0) ? tanhf(val) : 1.0f / (1.0f + __expf(-val));
        gatesLds[(g * 16 + M) * 32 + nl] = val;
    }
    __syncthreads();

    // Elementwise c/h update: 16 rows x 32 cols = 512 elems, 2 per thread.
#pragma unroll
    for (int e = 0; e < 2; ++e) {
        int idx = threadIdx.x + e * 256;
        int M   = idx >> 5;
        int nl2 = idx & 31;
        int hr  = hb * 16 + M;
        int n2  = nb2 * 32 + nl2;
        float gv = gatesLds[(0 * 16 + M) * 32 + nl2];
        float iv = gatesLds[(1 * 16 + M) * 32 + nl2];
        float fv = gatesLds[(2 * 16 + M) * 32 + nl2];
        float ov = gatesLds[(3 * 16 + M) * 32 + nl2];
        float cOld = c[hr * 512 + n2];
        float cNew = gv * iv + cOld * fv;
        c[hr * 512 + n2] = cNew;
        float hv = tanhf(cNew) * ov;
        // Store h already swizzled as next step's B operand (32x16 tile):
        // lane = 16*((K/8)%2) + n%16, half j = K%8 + 8*(K/16), K = hr%32.
        int K_in  = hr & 31;
        int laneS = ((K_in >> 3) & 1) * 16 + (n2 & 15);
        int j     = (K_in & 7) + ((K_in >> 4) & 1) * 8;
        h_dst[((hr >> 5) * 32 + (n2 >> 4)) * 512 + laneS * 16 + j] =
            f32_to_bf16(hv);
    }
}

// Final projection: p = W_ph @ h + b_p, output transposed to [B, 128].
// Grid (8, 8), 128 threads (4 waves); wave handles one 16-col batch block.
__global__ __launch_bounds__(128) void lstm_proj(
    const unsigned short* __restrict__ Wph_sw,
    const unsigned short* __restrict__ h_sw,
    const float* __restrict__ b_p,
    float* __restrict__ out)
{
    const int mb   = blockIdx.x;                  // 0..7 (class rows /16)
    const int wave = threadIdx.x >> 5;
    const int lane = threadIdx.x & 31;
    const int nblk = blockIdx.y * 4 + wave;       // 0..31

    const unsigned short* aPtr = Wph_sw + mb * 16 * 512 + lane * 16;
    const unsigned short* bPtr = h_sw + nblk * 512 + lane * 16;

    v8f acc = {};
#pragma unroll
    for (int kb = 0; kb < 16; ++kb) {
        v16bf a = *reinterpret_cast<const v16bf*>(aPtr + kb * 512);
        v16bf b = *reinterpret_cast<const v16bf*>(bPtr + kb * (32 * 512));
        acc = __builtin_amdgcn_wmma_f32_16x16x32_bf16(
            false, a, false, b, (short)0, acc, false, false);
    }
    union { v8f v; float f[8]; } u;
    u.v = acc;

    int n = nblk * 16 + (lane & 15);
#pragma unroll
    for (int r = 0; r < 8; ++r) {
        int m = mb * 16 + 8 * (lane >> 4) + r;
        out[n * 128 + m] = u.f[r] + b_p[m];
    }
}

extern "C" void kernel_launch(void* const* d_in, const int* in_sizes, int n_in,
                              void* d_out, int out_size, void* d_ws, size_t ws_size,
                              hipStream_t stream) {
    (void)in_sizes; (void)n_in; (void)out_size; (void)ws_size;
    const float* x    = (const float*)d_in[0];
    const float* W_x  = (const float*)d_in[1];
    const float* W_h  = (const float*)d_in[2];
    const float* W_ph = (const float*)d_in[3];
    const float* b_p  = (const float*)d_in[4];
    const float* bg   = (const float*)d_in[5];
    const float* bi   = (const float*)d_in[6];
    const float* bf_  = (const float*)d_in[7];
    const float* bo   = (const float*)d_in[8];
    float* out = (float*)d_out;

    char* ws = (char*)d_ws;
    unsigned short* Wh_sw  = (unsigned short*)(ws);
    unsigned short* Wph_sw = (unsigned short*)(ws + 2097152);
    unsigned short* h0     = (unsigned short*)(ws + 2228224);
    unsigned short* h1     = (unsigned short*)(ws + 2752512);
    float*          c      = (float*)        (ws + 3276800);

    // One-time (per call) weight conversion + state init — all on `stream`,
    // graph-capture safe.
    convert_swizzle_A<<<4096, 256, 0, stream>>>(W_h, Wh_sw, 128);
    convert_swizzle_A<<<256, 256, 0, stream>>>(W_ph, Wph_sw, 8);
    (void)hipMemsetAsync(h0, 0, 524288, stream);
    (void)hipMemsetAsync(c, 0, 1048576, stream);

    // torch loop: timesteps 0 .. SEQ_LEN-2  (127 steps), ping-pong h buffers.
    for (int t = 0; t < 127; ++t) {
        const unsigned short* src = (t & 1) ? h1 : h0;
        unsigned short*       dst = (t & 1) ? h0 : h1;
        lstm_step<<<dim3(32, 16), 256, 0, stream>>>(
            Wh_sw, x, W_x, bg, bi, bf_, bo, src, dst, c, t);
    }
    // 127 steps end with h in h1.
    lstm_proj<<<dim3(8, 8), 128, 0, stream>>>(Wph_sw, h1, b_p, out);
}
